// ConvLayer_30004641530003
// MI455X (gfx1250) — matrix operand order repaired and checked by reading, not compiled, not run on hardware
//
#include <hip/hip_runtime.h>
#include <math.h>

#define EPS 1e-5f
#define NNODES 50000
#define NEDGES 512000
#define FDIM 128

typedef __attribute__((ext_vector_type(16))) __bf16 v16bf;
typedef __attribute__((ext_vector_type(8)))  __bf16 v8bf;
typedef __attribute__((ext_vector_type(8)))  float  v8f;

// ---------------------------------------------------------------------------
// Column stats for the gathered edge concat matrix [ae0 | ae1 | edge] (384 cols)
// sums layout: [0..383]=sum, [384..767]=sumsq
// ---------------------------------------------------------------------------
__global__ void edge_cat_stats_kernel(const float* __restrict__ atom,
                                      const float* __restrict__ edge,
                                      const int* __restrict__ idx,
                                      long E, float* __restrict__ sums) {
  int t = threadIdx.x;  // 0..127
  float s0 = 0.f, q0 = 0.f, s1 = 0.f, q1 = 0.f, s2 = 0.f, q2 = 0.f;
  for (long r = blockIdx.x; r < E; r += gridDim.x) {
    int i0 = idx[2 * r], i1 = idx[2 * r + 1];
    float a0 = atom[(long)i0 * FDIM + t];
    float a1 = atom[(long)i1 * FDIM + t];
    float ee = edge[r * FDIM + t];
    s0 += a0; q0 += a0 * a0;
    s1 += a1; q1 += a1 * a1;
    s2 += ee; q2 += ee * ee;
  }
  atomicAdd(&sums[t],             s0); atomicAdd(&sums[384 + t],       q0);
  atomicAdd(&sums[128 + t],       s1); atomicAdd(&sums[384 + 128 + t], q1);
  atomicAdd(&sums[256 + t],       s2); atomicAdd(&sums[384 + 256 + t], q2);
}

// Column stats of an M x 128 matrix, written at column offset into a 2*outC buf
__global__ void col_stats128_kernel(const float* __restrict__ x, long M,
                                    float* __restrict__ sums, int outC, int colOff) {
  int t = threadIdx.x;
  float s = 0.f, q = 0.f;
  for (long r = blockIdx.x; r < M; r += gridDim.x) {
    float v = x[r * FDIM + t];
    s += v; q += v * v;
  }
  atomicAdd(&sums[colOff + t], s);
  atomicAdd(&sums[outC + colOff + t], q);
}

// ---------------------------------------------------------------------------
// Compose bn2∘bn1 (cols in [compLo,compHi)) or bn2 alone into per-column affine
// ---------------------------------------------------------------------------
__global__ void cat_affine_kernel(const float* __restrict__ sums, int C,
                                  int compLo, int compHi,
                                  const float* __restrict__ g1,
                                  const float* __restrict__ b1,
                                  const float* __restrict__ g2,
                                  const float* __restrict__ b2,
                                  float cnt, float* __restrict__ alpha,
                                  float* __restrict__ beta) {
  int c = blockIdx.x * blockDim.x + threadIdx.x;
  if (c >= C) return;
  float m = sums[c] / cnt;
  float v = fmaxf(sums[C + c] / cnt - m * m, 0.f);
  float al, be;
  if (c >= compLo && c < compHi) {
    float G1 = g1[c - compLo];
    float r1 = rsqrtf(v + EPS);
    float vy = G1 * G1 * v * r1 * r1;   // variance after bn1 (shift b1 cancels)
    float r2 = rsqrtf(vy + EPS);
    al = G1 * r1 * g2[c] * r2;
    be = b2[c] - al * m;
  } else {
    float r = rsqrtf(v + EPS);
    al = g2[c] * r;
    be = b2[c] - al * m;
  }
  alpha[c] = al; beta[c] = be;
  (void)b1;
}

// Fold input-affine into weights: Wb[n][k] = bf16(W[n][k]*alpha[k]),
// bprime[n] = lin_b[n] + sum_k W[n][k]*beta[k]
__global__ void fold_weights_kernel(const float* __restrict__ W,
                                    const float* __restrict__ lin_b,
                                    const float* __restrict__ alpha,
                                    const float* __restrict__ beta,
                                    int K, __bf16* __restrict__ Wb,
                                    float* __restrict__ bprime) {
  int n = blockIdx.x;   // 0..127
  int t = threadIdx.x;  // 0..127
  __shared__ float red[128];
  float dot = 0.f;
  for (int k = t; k < K; k += 128) {
    float w = W[(long)n * K + k];
    Wb[(long)n * K + k] = (__bf16)(w * alpha[k]);
    dot += w * beta[k];
  }
  red[t] = dot;
  __syncthreads();
  for (int s = 64; s > 0; s >>= 1) {
    if (t < s) red[t] += red[t + s];
    __syncthreads();
  }
  if (t == 0) bprime[n] = lin_b[n] + red[0];
}

// Per-column affine from 128-col stats (used for dbn and bn3)
__global__ void make_affine128_kernel(const float* __restrict__ sums,
                                      const float* __restrict__ g,
                                      const float* __restrict__ b, float cnt,
                                      float* __restrict__ alpha,
                                      float* __restrict__ beta) {
  int c = threadIdx.x;
  float m = sums[c] / cnt;
  float v = fmaxf(sums[FDIM + c] / cnt - m * m, 0.f);
  float r = rsqrtf(v + EPS);
  float al = g[c] * r;
  alpha[c] = al; beta[c] = b[c] - al * m;
}

// z <- base + softplus(alpha*z + beta); accumulate stats of result
__global__ void residual_softplus_kernel(float* __restrict__ z,
                                         const float* __restrict__ base,
                                         const float* __restrict__ alpha,
                                         const float* __restrict__ beta, long M,
                                         float* __restrict__ sums) {
  int t = threadIdx.x;
  float al = alpha[t], be = beta[t];
  float s = 0.f, q = 0.f;
  for (long r = blockIdx.x; r < M; r += gridDim.x) {
    float x = al * z[r * FDIM + t] + be;
    float sp = fmaxf(x, 0.f) + log1pf(expf(-fabsf(x)));
    float e2 = base[r * FDIM + t] + sp;
    z[r * FDIM + t] = e2;
    s += e2; q += e2 * e2;
  }
  atomicAdd(&sums[t], s); atomicAdd(&sums[FDIM + t], q);
}

__global__ void apply_affine_kernel(const float* __restrict__ src,
                                    const float* __restrict__ alpha,
                                    const float* __restrict__ beta,
                                    float* __restrict__ dst, long M) {
  int t = threadIdx.x;
  float al = alpha[t], be = beta[t];
  for (long r = blockIdx.x; r < M; r += gridDim.x)
    dst[r * FDIM + t] = al * src[r * FDIM + t] + be;
}

__global__ void scatter_add_kernel(const float* __restrict__ edge,
                                   const int* __restrict__ idx, long E,
                                   float* __restrict__ psum,
                                   float* __restrict__ cnt) {
  int t = threadIdx.x;
  for (long r = blockIdx.x; r < E; r += gridDim.x) {
    int n = idx[2 * r];
    atomicAdd(&psum[(long)n * FDIM + t], edge[r * FDIM + t]);
    if (t == 0) atomicAdd(&cnt[n], 1.f);
  }
}

__global__ void pooled_div_kernel(float* __restrict__ pooled,
                                  const float* __restrict__ cnt, long N) {
  int t = threadIdx.x;
  for (long r = blockIdx.x; r < N; r += gridDim.x) {
    float c = fmaxf(cnt[r], 1.f);
    pooled[r * FDIM + t] *= (1.f / c);
  }
}

// ---------------------------------------------------------------------------
// WMMA GEMM: Z[M x 128] = cat[M x K] @ Wb^T + bprime
// mode 0 (edge): cat cols 0..127 = atom[idx0], 128..255 = atom[idx1],
//                256..383 = edge row                         (K = 384)
// mode 1 (atom): cat cols 0..127 = atom row, 128..255 = pooled row (K = 256)
// Block tile 64(M) x 128(N), 8 wave32 waves; wave -> 16-row strip x 64-col half.
// Ping-pong LDS A tile (1 barrier / K-chunk) + register prefetch of the next
// gathered chunk so global latency overlaps the WMMAs.
// ---------------------------------------------------------------------------
__global__ __launch_bounds__(256) void gemm_cat_kernel(
    const float* __restrict__ A0, const float* __restrict__ A1,
    const int* __restrict__ idx, const __bf16* __restrict__ Wb,
    const float* __restrict__ bprime, float* __restrict__ Z, long M, int K,
    int mode) {
  __shared__ __bf16 Alds[2][64][32];  // 8 KB ping-pong bf16 A tile

  int tid = threadIdx.x;
  int wave = tid >> 5, lane = tid & 31;
  int mstrip = wave >> 1;  // 0..3 -> 16-row strips
  int nhalf = wave & 1;    // 0..1 -> 64-col halves
  long rowbase = (long)blockIdx.x * 64;

  v8f acc[4];
#pragma unroll
  for (int j = 0; j < 4; ++j)
#pragma unroll
    for (int r = 0; r < 8; ++r) acc[j][r] = 0.f;

  int lrow = tid >> 2;        // 0..63
  int lcol = (tid & 3) * 8;   // 0,8,16,24
  long gr = rowbase + lrow;
  bool rowok = (gr < M);

  // Row base pointers per 128-col segment (indices hoisted out of K loop)
  const float* bases[3];
  if (rowok) {
    if (mode == 0) {
      bases[0] = A0 + (long)idx[2 * gr] * FDIM;
      bases[1] = A0 + (long)idx[2 * gr + 1] * FDIM;
      bases[2] = A1 + gr * FDIM;
    } else {
      bases[0] = A0 + gr * FDIM;
      bases[1] = A1 + gr * FDIM;
      bases[2] = bases[1];
    }
  } else {
    bases[0] = bases[1] = bases[2] = A0;  // unused values (zeros substituted)
  }

  auto loadChunk = [&](int kk, float* vals) {
    int kloc = kk + lcol;
    const float* s = bases[kloc >> 7] + (kloc & 127);
    if (rowok) {
#pragma unroll
      for (int i = 0; i < 8; ++i) vals[i] = s[i];
    } else {
#pragma unroll
      for (int i = 0; i < 8; ++i) vals[i] = 0.f;
    }
  };

  float vals[8];
  loadChunk(0, vals);
  int nchunks = K >> 5;

  for (int c = 0; c < nchunks; ++c) {
    int buf = c & 1;
    // stage current chunk (fp32 -> bf16) into ping-pong LDS buffer
#pragma unroll
    for (int i = 0; i < 8; ++i) Alds[buf][lrow][lcol + i] = (__bf16)vals[i];
    // issue the gather for the next chunk before the barrier so its VMEM
    // latency overlaps this chunk's WMMAs
    if (c + 1 < nchunks) loadChunk((c + 1) << 5, vals);
    __syncthreads();

    // ---- A fragment (contiguous k = 16*(lane/16) + s, matching B) ----
    int m = mstrip * 16 + (lane & 15);
    int kb = (lane >> 4) * 16;
    v8bf lo = *(const v8bf*)&Alds[buf][m][kb];
    v8bf hi = *(const v8bf*)&Alds[buf][m][kb + 8];
    v16bf a;
#pragma unroll
    for (int i = 0; i < 8; ++i) { a[i] = lo[i]; a[8 + i] = hi[i]; }

    // ---- B fragments direct from L2-resident bf16 weights ----
    int kb2 = (c << 5) + (lane >> 4) * 16;
#pragma unroll
    for (int j = 0; j < 4; ++j) {
      int ncol = (nhalf * 4 + j) * 16 + (lane & 15);
      v16bf b = *(const v16bf*)(Wb + (long)ncol * K + kb2);
      acc[j] = __builtin_amdgcn_wmma_f32_16x16x32_bf16(
          false, a, false, b, (short)0, acc[j], false, false);
    }
  }

  // ---- epilogue: D layout -> global (add folded bias) ----
  int nlane = lane & 15;
  int mhi = (lane >> 4) * 8;
#pragma unroll
  for (int j = 0; j < 4; ++j) {
    int ncol = (nhalf * 4 + j) * 16 + nlane;
    float bb = bprime[ncol];
#pragma unroll
    for (int r = 0; r < 8; ++r) {
      long mg = rowbase + mstrip * 16 + mhi + r;
      if (mg < M) Z[mg * FDIM + ncol] = acc[j][r] + bb;
    }
  }
}

// ---------------------------------------------------------------------------
extern "C" void kernel_launch(void* const* d_in, const int* in_sizes, int n_in,
                              void* d_out, int out_size, void* d_ws,
                              size_t ws_size, hipStream_t stream) {
  (void)in_sizes; (void)n_in; (void)out_size; (void)ws_size;
  const long E = NEDGES, N = NNODES;
  const float* atom_in = (const float*)d_in[0];
  const float* edge_in = (const float*)d_in[1];
  const int* idx = (const int*)d_in[2];
  auto P = [&](int i) { return (const float*)d_in[i]; };
  // e-layer i (1..3): base 3+(i-1)*10 ; a-layer i (1..2): base 33+(i-1)*10
  // offsets: 0 bn1_g 1 bn1_b 2 bn2_g 3 bn2_b 4 bn3_g 5 bn3_b 6 lin_w 7 lin_b 8 dbn_g 9 dbn_b

  char* ws = (char*)d_ws;
  size_t off = 0;
  auto alloc = [&](size_t bytes) -> void* {
    void* p = ws + off;
    off = (off + bytes + 255) & ~(size_t)255;
    return p;
  };
  float* edge_cur = (float*)alloc(E * FDIM * 4);
  float* edge_tmp = (float*)alloc(E * FDIM * 4);
  float* atom_cur = (float*)alloc(N * FDIM * 4);
  float* atom_tmp = (float*)alloc(N * FDIM * 4);
  float* pooled   = (float*)alloc(N * FDIM * 4);
  float* counts   = (float*)alloc(N * 4);
  __bf16* Wb      = (__bf16*)alloc(128 * 384 * 2);
  float* bprime   = (float*)alloc(128 * 4);
  float* sums_raw = (float*)alloc(2 * 384 * 4);
  float* sums_z   = (float*)alloc(2 * 128 * 4);
  float* sums_e2  = (float*)alloc(2 * 128 * 4);
  float* alphaK   = (float*)alloc(384 * 4);
  float* betaK    = (float*)alloc(384 * 4);
  float* alpha1   = (float*)alloc(128 * 4);
  float* beta1    = (float*)alloc(128 * 4);
  float* alpha2   = (float*)alloc(128 * 4);
  float* beta2    = (float*)alloc(128 * 4);

  hipMemcpyAsync(edge_cur, edge_in, E * FDIM * 4, hipMemcpyDeviceToDevice, stream);
  hipMemcpyAsync(atom_cur, atom_in, N * FDIM * 4, hipMemcpyDeviceToDevice, stream);

  const int GS = 2048;                    // grid-stride blocks, 128 thr each
  const int GEMM_E = (int)((E + 63) / 64);
  const int GEMM_A = (int)((N + 63) / 64);

  for (int i = 1; i <= 3; ++i) {
    int eb = 3 + (i - 1) * 10;
    // --- edge block ---
    hipMemsetAsync(sums_raw, 0, 2 * 384 * 4, stream);
    edge_cat_stats_kernel<<<GS, 128, 0, stream>>>(atom_cur, edge_cur, idx, E, sums_raw);
    cat_affine_kernel<<<3, 128, 0, stream>>>(sums_raw, 384, 0, 256, P(eb + 0),
                                             P(eb + 1), P(eb + 2), P(eb + 3),
                                             (float)E, alphaK, betaK);
    fold_weights_kernel<<<128, 128, 0, stream>>>(P(eb + 6), P(eb + 7), alphaK,
                                                 betaK, 384, Wb, bprime);
    gemm_cat_kernel<<<GEMM_E, 256, 0, stream>>>(atom_cur, edge_cur, idx, Wb,
                                                bprime, edge_tmp, E, 384, 0);
    hipMemsetAsync(sums_z, 0, 2 * 128 * 4, stream);
    col_stats128_kernel<<<GS, 128, 0, stream>>>(edge_tmp, E, sums_z, 128, 0);
    make_affine128_kernel<<<1, 128, 0, stream>>>(sums_z, P(eb + 8), P(eb + 9),
                                                 (float)E, alpha1, beta1);
    hipMemsetAsync(sums_e2, 0, 2 * 128 * 4, stream);
    residual_softplus_kernel<<<GS, 128, 0, stream>>>(edge_tmp, edge_cur, alpha1,
                                                     beta1, E, sums_e2);
    make_affine128_kernel<<<1, 128, 0, stream>>>(sums_e2, P(eb + 4), P(eb + 5),
                                                 (float)E, alpha2, beta2);
    float* edst = (i == 3) ? (float*)d_out : edge_cur;
    apply_affine_kernel<<<GS, 128, 0, stream>>>(edge_tmp, alpha2, beta2, edst, E);

    if (i == 3) break;
    // --- atom block ---
    int ab = 33 + (i - 1) * 10;
    hipMemsetAsync(pooled, 0, N * FDIM * 4, stream);
    hipMemsetAsync(counts, 0, N * 4, stream);
    scatter_add_kernel<<<GS, 128, 0, stream>>>(edge_cur, idx, E, pooled, counts);
    pooled_div_kernel<<<GS, 128, 0, stream>>>(pooled, counts, N);
    hipMemsetAsync(sums_raw, 0, 2 * 256 * 4, stream);
    col_stats128_kernel<<<GS, 128, 0, stream>>>(atom_cur, N, sums_raw, 256, 0);
    col_stats128_kernel<<<GS, 128, 0, stream>>>(pooled, N, sums_raw, 256, 128);
    cat_affine_kernel<<<2, 128, 0, stream>>>(sums_raw, 256, 128, 256, P(ab + 0),
                                             P(ab + 1), P(ab + 2), P(ab + 3),
                                             (float)N, alphaK, betaK);
    fold_weights_kernel<<<128, 128, 0, stream>>>(P(ab + 6), P(ab + 7), alphaK,
                                                 betaK, 256, Wb, bprime);
    gemm_cat_kernel<<<GEMM_A, 256, 0, stream>>>(atom_cur, pooled, idx, Wb,
                                                bprime, atom_tmp, N, 256, 1);
    hipMemsetAsync(sums_z, 0, 2 * 128 * 4, stream);
    col_stats128_kernel<<<GS, 128, 0, stream>>>(atom_tmp, N, sums_z, 128, 0);
    make_affine128_kernel<<<1, 128, 0, stream>>>(sums_z, P(ab + 8), P(ab + 9),
                                                 (float)N, alpha1, beta1);
    hipMemsetAsync(sums_e2, 0, 2 * 128 * 4, stream);
    residual_softplus_kernel<<<GS, 128, 0, stream>>>(atom_tmp, atom_cur, alpha1,
                                                     beta1, N, sums_e2);
    make_affine128_kernel<<<1, 128, 0, stream>>>(sums_e2, P(ab + 4), P(ab + 5),
                                                 (float)N, alpha2, beta2);
    apply_affine_kernel<<<GS, 128, 0, stream>>>(atom_tmp, alpha2, beta2,
                                                atom_cur, N);
  }
}
